// MCDroupoutLSTM_42073499631872
// MI455X (gfx1250) — compile-verified, matrix operand order checked by reading, not run on hardware
//
#include <hip/hip_runtime.h>

typedef __attribute__((ext_vector_type(16))) _Float16 v16h;
typedef __attribute__((ext_vector_type(8)))  _Float16 v8h;
typedef __attribute__((ext_vector_type(8)))  float    v8f;

namespace {

constexpr int HIDDEN   = 64;
constexpr int NG       = 256;   // 4*HIDDEN gate rows
constexpr int OBS_LEN  = 20;
constexpr int PRED_LEN = 12;
constexpr int WAVES    = 8;
constexpr int ROWS_PER_BLOCK = WAVES * 16;

// Augmented K (multiple of 32):  layer0/dec: [h(64) | x(2) | pad] ; layer1: [h1(64) | h0(64)]
constexpr int KA0 = 96;
constexpr int KA1 = 128;
constexpr int KAD = 96;

// LDS byte layout (dynamic shared)
constexpr int OFF_WA0  = 0;                     // f16[256*96]
constexpr int OFF_WA1  = OFF_WA0 + NG*KA0*2;    // 49152, f16[256*128]
constexpr int OFF_WAD  = OFF_WA1 + NG*KA1*2;    // 114688, f16[256*96]
constexpr int OFF_BS   = OFF_WAD + NG*KAD*2;    // 163840, float[3*256] summed biases
constexpr int OFF_OW   = OFF_BS + 3*NG*4;       // 166912, float[128] out_w
constexpr int OFF_OB   = OFF_OW + 128*4;        // 167424, float[2]  out_b
constexpr int OFF_WAVE = 167936;                // 16B aligned
constexpr int WAVE_STRIDE = 9216;               // h0: 2*2048, h1: 2*2048, xb: 1024
constexpr int SMEM_TOTAL  = OFF_WAVE + WAVES*WAVE_STRIDE;  // 241664 B (< 320KB/WGP)

__device__ __forceinline__ float fast_sigmoid(float x) {
  return __builtin_amdgcn_rcpf(1.0f + __expf(-x));
}
__device__ __forceinline__ float fast_tanh(float x) {
  return 1.0f - 2.0f * __builtin_amdgcn_rcpf(__expf(2.0f * x) + 1.0f);
}

// Load one 16x32 f16 operand fragment (A or B^T stored row-major (row, k)).
// ISA layout: lanes 0-15 hold row r, K {0..7,16..23}; lanes 16-31 hold row r, K {8..15,24..31}.
__device__ __forceinline__ v16h load_frag(const _Float16* base, int rowStride, int r, int hi) {
  const _Float16* p = base + r * rowStride + hi * 8;
  v8h lo = *(const v8h*)(p);
  v8h hh = *(const v8h*)(p + 16);
  v16h out;
#pragma unroll
  for (int i = 0; i < 8; ++i) { out[i] = lo[i]; out[i + 8] = hh[i]; }
  return out;
}

// One fused LSTM cell for a 16-row batch tile, gates via WMMA.
// Gate-group trick: tiles {g, g+4, g+8, g+12} put i/f/g/o for element (m,k)
// at the same (lane, vgpr) slot -> register-local activation + state update.
// Accumulators start from inline-zero C; bias is added as a scalar in the epilogue
// (avoids caching 32 splatted bias registers across the time loop).
template <int NKC>
__device__ __forceinline__ void lstm_cell(const _Float16* const (&asrc)[NKC],
                                          const int (&astride)[NKC],
                                          const _Float16* wa, int ka,
                                          const float* bs,
                                          _Float16* hnext,
                                          float (&c)[32], int r, int hi) {
#pragma unroll
  for (int g = 0; g < 4; ++g) {
    v8f acc[4];
    // ---- kc = 0: C comes from inline zero ----
    {
      v16h a = load_frag(asrc[0], astride[0], r, hi);
      v16h b[4];
#pragma unroll
      for (int ti = 0; ti < 4; ++ti)
        b[ti] = load_frag(wa + (size_t)(4 * ti + g) * 16 * ka, ka, r, hi);
      v8f z = {};
#pragma unroll
      for (int ti = 0; ti < 4; ++ti)
        acc[ti] = __builtin_amdgcn_wmma_f32_16x16x32_f16(
            false, a, false, b[ti], (short)0, z, false, false);
    }
    // ---- remaining K chunks ----
#pragma unroll
    for (int kc = 1; kc < NKC; ++kc) {
      v16h a = load_frag(asrc[kc], astride[kc], r, hi);
      v16h b[4];
#pragma unroll
      for (int ti = 0; ti < 4; ++ti)
        b[ti] = load_frag(wa + (size_t)(4 * ti + g) * 16 * ka + kc * 32, ka, r, hi);
#pragma unroll
      for (int ti = 0; ti < 4; ++ti)
        acc[ti] = __builtin_amdgcn_wmma_f32_16x16x32_f16(
            false, a, false, b[ti], (short)0, acc[ti], false, false);
    }
    // ---- epilogue: bias + activations + state update ----
    // element (m = v + 8*hi, k = g*16 + r); bias n = (4*ti+g)*16 + r
    float bi = bs[(g    ) * 16 + r];
    float bf = bs[(g + 4) * 16 + r];
    float bg = bs[(g + 8) * 16 + r];
    float bo = bs[(g + 12) * 16 + r];
#pragma unroll
    for (int v = 0; v < 8; ++v) {
      float iv = fast_sigmoid(acc[0][v] + bi);
      float fv = fast_sigmoid(acc[1][v] + bf);
      float gv = fast_tanh(acc[2][v] + bg);
      float ov = fast_sigmoid(acc[3][v] + bo);
      float cn = fv * c[g * 8 + v] + iv * gv;
      c[g * 8 + v] = cn;
      hnext[(size_t)(v + 8 * hi) * HIDDEN + g * 16 + r] = (_Float16)(ov * fast_tanh(cn));
    }
  }
}

__global__ __launch_bounds__(WAVES * 32, 1)
void lstm_traj_kernel(const float* __restrict__ obs,
                      const float* __restrict__ w_ih_l0, const float* __restrict__ w_hh_l0,
                      const float* __restrict__ b_ih_l0, const float* __restrict__ b_hh_l0,
                      const float* __restrict__ w_ih_l1, const float* __restrict__ w_hh_l1,
                      const float* __restrict__ b_ih_l1, const float* __restrict__ b_hh_l1,
                      const float* __restrict__ dec_w_ih, const float* __restrict__ dec_w_hh,
                      const float* __restrict__ dec_b_ih, const float* __restrict__ dec_b_hh,
                      const float* __restrict__ out_w, const float* __restrict__ out_b,
                      float* __restrict__ out) {
  extern __shared__ __align__(16) char smem[];
  _Float16* WA0 = (_Float16*)(smem + OFF_WA0);
  _Float16* WA1 = (_Float16*)(smem + OFF_WA1);
  _Float16* WAD = (_Float16*)(smem + OFF_WAD);
  float* BS  = (float*)(smem + OFF_BS);
  float* OW  = (float*)(smem + OFF_OW);
  float* OBv = (float*)(smem + OFF_OB);

  const int tid = threadIdx.x;

  // ---- cooperative weight staging into LDS (augmented gate matrices, f16) ----
  for (int idx = tid; idx < NG * KA0; idx += WAVES * 32) {
    int n = idx / KA0, k = idx - n * KA0;
    float v = (k < 64) ? w_hh_l0[n * 64 + k] : ((k < 66) ? w_ih_l0[n * 2 + (k - 64)] : 0.0f);
    WA0[idx] = (_Float16)v;
  }
  for (int idx = tid; idx < NG * KA1; idx += WAVES * 32) {
    int n = idx / KA1, k = idx - n * KA1;
    float v = (k < 64) ? w_hh_l1[n * 64 + k] : w_ih_l1[n * 64 + (k - 64)];
    WA1[idx] = (_Float16)v;
  }
  for (int idx = tid; idx < NG * KAD; idx += WAVES * 32) {
    int n = idx / KAD, k = idx - n * KAD;
    float v = (k < 64) ? dec_w_hh[n * 64 + k] : ((k < 66) ? dec_w_ih[n * 2 + (k - 64)] : 0.0f);
    WAD[idx] = (_Float16)v;
  }
  for (int idx = tid; idx < NG; idx += WAVES * 32) {
    BS[idx]          = b_ih_l0[idx] + b_hh_l0[idx];
    BS[NG + idx]     = b_ih_l1[idx] + b_hh_l1[idx];
    BS[2 * NG + idx] = dec_b_ih[idx] + dec_b_hh[idx];
  }
  if (tid < 128) OW[tid] = out_w[tid];
  if (tid < 2)   OBv[tid] = out_b[tid];

  // ---- per-wave state buffers ----
  const int wave = tid >> 5, lane = tid & 31;
  const int r = lane & 15, hi = lane >> 4;
  char* wb = smem + OFF_WAVE + wave * WAVE_STRIDE;
  _Float16* H0 = (_Float16*)(wb);           // two ping-pong 16x64 f16 buffers
  _Float16* H1 = (_Float16*)(wb + 4096);
  _Float16* XB = (_Float16*)(wb + 8192);    // 16x32 f16, cols>=2 stay zero

  for (int i = lane; i < 1024; i += 32) {
    H0[i] = (_Float16)0.f; H0[1024 + i] = (_Float16)0.f;
    H1[i] = (_Float16)0.f; H1[1024 + i] = (_Float16)0.f;
  }
  for (int i = lane; i < 512; i += 32) XB[i] = (_Float16)0.f;

  __syncthreads();

  float c0[32], c1[32];
#pragma unroll
  for (int i = 0; i < 32; ++i) { c0[i] = 0.f; c1[i] = 0.f; }

  const int rowBase = blockIdx.x * ROWS_PER_BLOCK + wave * 16;
  const int xm = lane >> 1, xd = lane & 1;  // lane -> (batch row, coord) for x/out
  const size_t obsRow = (size_t)(rowBase + xm) * (OBS_LEN * 2);

  int p = 0;
  // ---- encoder: fused layer0 + layer1 per timestep ----
  for (int t = 0; t < OBS_LEN; ++t) {
    XB[xm * 32 + xd] = (_Float16)obs[obsRow + t * 2 + xd];
    {
      const _Float16* const a0[3] = { H0 + p * 1024, H0 + p * 1024 + 32, XB };
      const int s0[3] = {64, 64, 32};
      lstm_cell<3>(a0, s0, WA0, KA0, BS, H0 + (p ^ 1) * 1024, c0, r, hi);
    }
    {
      const _Float16* const a1[4] = { H1 + p * 1024, H1 + p * 1024 + 32,
                                      H0 + (p ^ 1) * 1024, H0 + (p ^ 1) * 1024 + 32 };
      const int s1[4] = {64, 64, 64, 64};
      lstm_cell<4>(a1, s1, WA1, KA1, BS + NG, H1 + (p ^ 1) * 1024, c1, r, hi);
    }
    p ^= 1;
  }

  // ---- decoder: autoregressive, continues (h1, c1) ----
  XB[xm * 32 + xd] = (_Float16)obs[obsRow + (OBS_LEN - 1) * 2 + xd];
  for (int s = 0; s < PRED_LEN; ++s) {
    {
      const _Float16* const ad[3] = { H1 + p * 1024, H1 + p * 1024 + 32, XB };
      const int sd[3] = {64, 64, 32};
      lstm_cell<3>(ad, sd, WAD, KAD, BS + 2 * NG, H1 + (p ^ 1) * 1024, c1, r, hi);
    }
    p ^= 1;
    // out = h @ out_w.T + out_b ; one (m, d) element per lane, vectorized reads
    const _Float16* h = H1 + p * 1024;
    float acc = OBv[xd];
#pragma unroll
    for (int kk = 0; kk < 8; ++kk) {
      v8h hv = *(const v8h*)(h + xm * 64 + kk * 8);
      const float4* owp = (const float4*)(OW + xd * 64 + kk * 8);
      float4 w0 = owp[0], w1 = owp[1];
      acc += (float)hv[0] * w0.x + (float)hv[1] * w0.y +
             (float)hv[2] * w0.z + (float)hv[3] * w0.w +
             (float)hv[4] * w1.x + (float)hv[5] * w1.y +
             (float)hv[6] * w1.z + (float)hv[7] * w1.w;
    }
    out[(size_t)(rowBase + xm) * (PRED_LEN * 2) + s * 2 + xd] = acc;
    XB[xm * 32 + xd] = (_Float16)acc;   // feed back as next input
  }
}

}  // namespace

extern "C" void kernel_launch(void* const* d_in, const int* in_sizes, int n_in,
                              void* d_out, int out_size, void* d_ws, size_t ws_size,
                              hipStream_t stream) {
  (void)n_in; (void)out_size; (void)d_ws; (void)ws_size;
  const float* obs      = (const float*)d_in[0];
  const float* w_ih_l0  = (const float*)d_in[1];
  const float* w_hh_l0  = (const float*)d_in[2];
  const float* b_ih_l0  = (const float*)d_in[3];
  const float* b_hh_l0  = (const float*)d_in[4];
  const float* w_ih_l1  = (const float*)d_in[5];
  const float* w_hh_l1  = (const float*)d_in[6];
  const float* b_ih_l1  = (const float*)d_in[7];
  const float* b_hh_l1  = (const float*)d_in[8];
  const float* dec_w_ih = (const float*)d_in[9];
  const float* dec_w_hh = (const float*)d_in[10];
  const float* dec_b_ih = (const float*)d_in[11];
  const float* dec_b_hh = (const float*)d_in[12];
  const float* out_w    = (const float*)d_in[13];
  const float* out_b    = (const float*)d_in[14];

  const int B = in_sizes[0] / (OBS_LEN * 2);     // 32768
  const int grid = B / ROWS_PER_BLOCK;           // 256 blocks of 8 waves

  (void)hipFuncSetAttribute((const void*)lstm_traj_kernel,
                            hipFuncAttributeMaxDynamicSharedMemorySize, SMEM_TOTAL);

  lstm_traj_kernel<<<grid, WAVES * 32, SMEM_TOTAL, stream>>>(
      obs, w_ih_l0, w_hh_l0, b_ih_l0, b_hh_l0,
      w_ih_l1, w_hh_l1, b_ih_l1, b_hh_l1,
      dec_w_ih, dec_w_hh, dec_b_ih, dec_b_hh,
      out_w, out_b, (float*)d_out);
}
